// BWGNN_26542897889793
// MI455X (gfx1250) — matrix operand order repaired and checked by reading, not compiled, not run on hardware
//
#include <hip/hip_runtime.h>
#include <hip/hip_bf16.h>

typedef __attribute__((ext_vector_type(16))) __bf16 v16bf;
typedef __attribute__((ext_vector_type(8)))  float  v8f;

// ---------------- utility kernels ----------------

__global__ void zero_kernel(float* __restrict__ p, long n) {
  long t = (long)blockIdx.x * blockDim.x + threadIdx.x;
  if (t < n) p[t] = 0.0f;
}

__global__ void deg_kernel(const int* __restrict__ dst, float* __restrict__ deg, int E) {
  int e = blockIdx.x * blockDim.x + threadIdx.x;
  if (e < E) atomicAdd(deg + dst[e], 1.0f);
}

__global__ void dinv_kernel(float* __restrict__ dinv, int N) {
  int i = blockIdx.x * blockDim.x + threadIdx.x;
  if (i < N) dinv[i] = rsqrtf(fmaxf(dinv[i], 1.0f));
}

// ---------------- weight pre-pack: fp32 [64,K] -> bf16 WMMA B fragments ------
// Fragment (kc, j) covers K rows [kc*32, kc*32+32) x columns [j*16, j*16+16).
// Lane l of fragment holds 16 consecutive bf16: W[n][kbase .. kbase+15] where
// n = j*16 + (l&15), kbase = kc*32 + (l>>4)*16  (contiguous k order, same as the A fragment:
// lanes 0-15 = K 0..15, lanes 16-31 = K 16..31, column n = lane&15).
// Storage: Wp[((kc*4 + j)*32 + lane)*16 + t], lane-major 32-byte chunks.

__global__ void packW_kernel(const float* __restrict__ W, __bf16* __restrict__ Wp, int K) {
  int idx = blockIdx.x * blockDim.x + threadIdx.x;
  int total = (K >> 5) * 128;                 // (K/32) kc-chunks * 4 j-blocks * 32 lanes
  if (idx >= total) return;
  int lane = idx & 31;
  int j    = (idx >> 5) & 3;
  int kc   = idx >> 7;
  int n     = j * 16 + (lane & 15);
  int kbase = kc * 32 + (lane >> 4) * 16;
  const float* wr = W + (size_t)n * K + kbase;
  __bf16* dp = Wp + (size_t)idx * 16;
#pragma unroll
  for (int t = 0; t < 16; ++t) dp[t] = (__bf16)wr[t];
}

// ---------------- WMMA GEMM: Y[nrows,64] = relu(X[nrows,K] @ W[64,K]^T + b) ----
// One wave computes a 16x64 tile via 4 accumulators of v_wmma_f32_16x16x32_bf16.
// A-fragment (16-bit A 16x32): lanes 0-15 = rows M with K 0..15 (contiguous);
// lanes 16-31 = rows M with K 16..31; converted fp32->bf16 on the fly.
// B fragments come pre-packed in bf16 (two b128 loads each, no cvt).
// C/D (f32 16x16): lane n = lane&15, VGPR v -> row M = (lane>>4)*8 + v.

__global__ __launch_bounds__(256)
void gemm_relu_wmma(const float* __restrict__ X, int ldx,
                    const __bf16* __restrict__ Wp,    // packed B fragments
                    const float* __restrict__ bias,   // [64]
                    float* __restrict__ Y, int ldy,
                    int nrows, int K, int do_relu)
{
  const int lane  = threadIdx.x & 31;
  const int wave  = blockIdx.x * (blockDim.x >> 5) + (threadIdx.x >> 5);
  const int ntile = (nrows + 15) >> 4;
  if (wave >= ntile) return;                 // wave-uniform guard
  const int m0   = wave << 4;
  const int nlo  = lane & 15;
  const int half = lane >> 4;                // 0 or 1

  v8f acc0 = {}, acc1 = {}, acc2 = {}, acc3 = {};

  int arow = m0 + nlo;
  if (arow >= nrows) arow = nrows - 1;       // clamp (keeps EXEC all-1)

  const v16bf* Bfrag = (const v16bf*)Wp;

  for (int k0 = 0, kc = 0; k0 < K; k0 += 32, ++kc) {
    // ---- A fragment: row `arow`, contiguous K-halves, same order as packed B ----
    v16bf a;
    const float* xr = X + (size_t)arow * ldx + k0 + half * 16;
#pragma unroll
    for (int i = 0; i < 4; ++i) {
      float2 p = *(const float2*)(xr + 2 * i);        // K = half*16 + 2i
      a[2 * i]     = (__bf16)p.x;
      a[2 * i + 1] = (__bf16)p.y;
      float2 q = *(const float2*)(xr + 8 + 2 * i);    // K = half*16 + 8 + 2i
      a[8 + 2 * i]     = (__bf16)q.x;
      a[8 + 2 * i + 1] = (__bf16)q.y;
    }
    // ---- pre-packed B fragments: straight b128 loads into WMMA layout ----
    v16bf b0 = Bfrag[(size_t)(kc * 4 + 0) * 32 + lane];
    v16bf b1 = Bfrag[(size_t)(kc * 4 + 1) * 32 + lane];
    v16bf b2 = Bfrag[(size_t)(kc * 4 + 2) * 32 + lane];
    v16bf b3 = Bfrag[(size_t)(kc * 4 + 3) * 32 + lane];

    acc0 = __builtin_amdgcn_wmma_f32_16x16x32_bf16(false, a, false, b0, (short)0, acc0, false, false);
    acc1 = __builtin_amdgcn_wmma_f32_16x16x32_bf16(false, a, false, b1, (short)0, acc1, false, false);
    acc2 = __builtin_amdgcn_wmma_f32_16x16x32_bf16(false, a, false, b2, (short)0, acc2, false, false);
    acc3 = __builtin_amdgcn_wmma_f32_16x16x32_bf16(false, a, false, b3, (short)0, acc3, false, false);
  }

  // ---- epilogue: bias + relu + store ----
#pragma unroll
  for (int j = 0; j < 4; ++j) {
    v8f acc = (j == 0) ? acc0 : (j == 1) ? acc1 : (j == 2) ? acc2 : acc3;
    const int n  = j * 16 + nlo;
    const float bn = bias[n];
#pragma unroll
    for (int v = 0; v < 8; ++v) {
      int m = m0 + half * 8 + v;
      if (m < nrows) {
        float r = acc[v] + bn;
        if (do_relu) r = fmaxf(r, 0.0f);
        Y[(size_t)m * ldy + n] = r;
      }
    }
  }
}

// ---------------- graph propagation ----------------

// hfin[:, tcol:tcol+64] = theta0 * h ; f = h
__global__ void poly_init_kernel(const float* __restrict__ h, float* __restrict__ f,
                                 float* __restrict__ hfin, int tcol, float theta0, int N) {
  long t = (long)blockIdx.x * blockDim.x + threadIdx.x;
  if (t >= (long)N * 64) return;
  int node = (int)(t >> 6);
  int c    = (int)(t & 63);
  float v = h[t];
  f[t] = v;
  hfin[(size_t)node * 192 + tcol + c] = theta0 * v;
}

// agg[dst] += f[src] * dinv[src] ; 16 threads/edge, float4 each
__global__ void edge_scatter_kernel(const int* __restrict__ src, const int* __restrict__ dst,
                                    const float* __restrict__ f, const float* __restrict__ dinv,
                                    float* __restrict__ agg, int E) {
  long t = (long)blockIdx.x * blockDim.x + threadIdx.x;
  if (t >= (long)E * 16) return;
  int e = (int)(t >> 4);
  int c = (int)((t & 15) << 2);
  int s = src[e];
  int d = dst[e];
  float sc = dinv[s];
  float4 v = *(const float4*)(f + (size_t)s * 64 + c);
  float* ap = agg + (size_t)d * 64 + c;
  atomicAdd(ap + 0, v.x * sc);
  atomicAdd(ap + 1, v.y * sc);
  atomicAdd(ap + 2, v.z * sc);
  atomicAdd(ap + 3, v.w * sc);
}

// f = f - agg*dinv ; hfin[:, tcol:+64] += theta_k * f
__global__ void poly_update_kernel(float* __restrict__ f, const float* __restrict__ agg,
                                   const float* __restrict__ dinv,
                                   float* __restrict__ hfin, int tcol, float th, int N) {
  long t = (long)blockIdx.x * blockDim.x + threadIdx.x;
  if (t >= (long)N * 64) return;
  int node = (int)(t >> 6);
  int c    = (int)(t & 63);
  float fv = f[t] - agg[t] * dinv[node];
  f[t] = fv;
  hfin[(size_t)node * 192 + tcol + c] += th * fv;
}

// out[i,:2] = h3[i,:] @ W4^T + b4  (C=2, trivial)
__global__ void out_head_kernel(const float* __restrict__ h3, const float* __restrict__ W4,
                                const float* __restrict__ b4, float* __restrict__ out, int N) {
  int i = blockIdx.x * blockDim.x + threadIdx.x;
  if (i >= N) return;
  float a0 = b4[0], a1 = b4[1];
  const float* hr = h3 + (size_t)i * 64;
#pragma unroll 8
  for (int c = 0; c < 64; ++c) {
    float v = hr[c];
    a0 += v * W4[c];
    a1 += v * W4[64 + c];
  }
  out[2 * i]     = a0;
  out[2 * i + 1] = a1;
}

// ---------------- host orchestration ----------------

extern "C" void kernel_launch(void* const* d_in, const int* in_sizes, int n_in,
                              void* d_out, int out_size, void* d_ws, size_t ws_size,
                              hipStream_t stream) {
  const float* in_feat = (const float*)d_in[0];
  const int*   src     = (const int*)d_in[1];
  const int*   dst     = (const int*)d_in[2];
  const float* W1 = (const float*)d_in[3];
  const float* b1 = (const float*)d_in[4];
  const float* W2 = (const float*)d_in[5];
  const float* b2 = (const float*)d_in[6];
  const float* W3 = (const float*)d_in[7];
  const float* b3 = (const float*)d_in[8];
  const float* W4 = (const float*)d_in[9];
  const float* b4 = (const float*)d_in[10];
  float* out = (float*)d_out;

  const int N = in_sizes[0] / 128;
  const int E = in_sizes[1];

  float* ws   = (float*)d_ws;
  float* dinv = ws;                        // [N]
  float* bufA = dinv + N;                  // [N,64] h1, then f
  float* h    = bufA + (size_t)N * 64;     // [N,64] post-W2 features
  float* agg  = h    + (size_t)N * 64;     // [N,64] scatter target, reused as h3
  float* hfin = agg  + (size_t)N * 64;     // [N,192] concat of 3 poly convs

  // packed bf16 weight fragments (32B-aligned region after hfin)
  size_t woff = (size_t)N * 385;
  woff = (woff + 63) & ~(size_t)63;        // 64-float (256B) alignment
  __bf16* W1p = (__bf16*)(ws + woff);      // (128/32)*4*32*16 = 8192 bf16
  __bf16* W2p = W1p + 8192;                // ( 64/32)*4*32*16 = 4096 bf16
  __bf16* W3p = W2p + 4096;                // (192/32)*4*32*16 = 12288 bf16

  const int T = 256;
  auto blocks = [](long n, int t) { return (int)((n + t - 1) / t); };

  // degree -> d^{-1/2}
  zero_kernel<<<blocks(N, T), T, 0, stream>>>(dinv, N);
  deg_kernel<<<blocks(E, T), T, 0, stream>>>(dst, dinv, E);
  dinv_kernel<<<blocks(N, T), T, 0, stream>>>(dinv, N);

  // one-shot weight packing fp32 -> bf16 WMMA fragments
  packW_kernel<<<blocks(4 * 128, T), T, 0, stream>>>(W1, W1p, 128);
  packW_kernel<<<blocks(2 * 128, T), T, 0, stream>>>(W2, W2p, 64);
  packW_kernel<<<blocks(6 * 128, T), T, 0, stream>>>(W3, W3p, 192);

  // dense front-end (bf16 WMMA, fp32 accumulate)
  const int ntile = (N + 15) / 16;
  const int gblk  = (ntile + 7) / 8;       // 8 waves / 256-thread block
  gemm_relu_wmma<<<gblk, 256, 0, stream>>>(in_feat, 128, W1p, b1, bufA, 64, N, 128, 1);
  gemm_relu_wmma<<<gblk, 256, 0, stream>>>(bufA,     64, W2p, b2, h,    64, N,  64, 1);

  // three Beta-wavelet polynomial convolutions, 2 propagation steps each
  const float THETAS[3][3] = {{3.0f, -3.0f, 0.75f}, {0.0f, 3.0f, -1.5f}, {0.0f, 0.0f, 0.75f}};
  const long NH  = (long)N * 64;
  const long E16 = (long)E * 16;
  for (int t = 0; t < 3; ++t) {
    poly_init_kernel<<<blocks(NH, T), T, 0, stream>>>(h, bufA, hfin, t * 64, THETAS[t][0], N);
    for (int k = 1; k < 3; ++k) {
      zero_kernel<<<blocks(NH, T), T, 0, stream>>>(agg, NH);
      edge_scatter_kernel<<<blocks(E16, T), T, 0, stream>>>(src, dst, bufA, dinv, agg, E);
      poly_update_kernel<<<blocks(NH, T), T, 0, stream>>>(bufA, agg, dinv, hfin, t * 64, THETAS[t][k], N);
    }
  }

  // dense back-end + classifier head
  gemm_relu_wmma<<<gblk, 256, 0, stream>>>(hfin, 192, W3p, b3, agg, 64, N, 192, 1);
  out_head_kernel<<<blocks(N, T), T, 0, stream>>>(agg, W4, b4, out, N);
}
